// HierarchicalAttention_36687610643081
// MI455X (gfx1250) — matrix operand order repaired and checked by reading, not compiled, not run on hardware
//
#include <hip/hip_runtime.h>
#include <hip/hip_bf16.h>

// ---------------- constants (match reference) ----------------
#define DIMC   1024
#define SEQC   2048
#define HEADSC 8
#define HDC    128      // head dim
#define BLKC   64       // attention block
#define NBC    32       // SEQ / BLK
#define BATCHC 2

typedef __bf16 bf16;
typedef __bf16 v16bf __attribute__((ext_vector_type(16)));
typedef __bf16 v8bf  __attribute__((ext_vector_type(8)));
typedef float  v8f   __attribute__((ext_vector_type(8)));

static __device__ inline v8f zero8() {
  v8f z;
#pragma unroll
  for (int i = 0; i < 8; ++i) z[i] = 0.0f;
  return z;
}

static __device__ inline v8f wmma_bf16(v16bf a, v16bf b, v8f c) {
  return __builtin_amdgcn_wmma_f32_16x16x32_bf16(
      /*neg_a=*/false, a, /*neg_b=*/false, b,
      /*c_mod=*/(short)0, c, /*reuse_a=*/false, /*reuse_b=*/false);
}

// A-fragment (16x32, bf16): lane (g = lane>>4, r = lane&15) holds row M=r.
// elems 0..15 -> K = kc*32 + g*16 + e
// `row` points at this lane's matrix row (row-major).
static __device__ inline v16bf load_a_frag(const bf16* row, int g, int kc) {
  const v8bf lo = *reinterpret_cast<const v8bf*>(row + kc * 32 + g * 16);
  const v8bf hi = *reinterpret_cast<const v8bf*>(row + kc * 32 + g * 16 + 8);
  v16bf a;
#pragma unroll
  for (int i = 0; i < 8; ++i) { a[i] = lo[i]; a[i + 8] = hi[i]; }
  return a;
}

// B-fragment (32x16, bf16) from memory laid out so that the lane's 16
// K-values (K = g*16 + e) are contiguous at `p`.
static __device__ inline v16bf load_b_frag(const bf16* p) {
  const v8bf lo = *reinterpret_cast<const v8bf*>(p);
  const v8bf hi = *reinterpret_cast<const v8bf*>(p + 8);
  v16bf b;
#pragma unroll
  for (int i = 0; i < 8; ++i) { b[i] = lo[i]; b[i + 8] = hi[i]; }
  return b;
}

// ---------------- f32 -> bf16 convert ----------------
__global__ void cvt_kernel(const float* __restrict__ src, bf16* __restrict__ dst, int n) {
  int i = blockIdx.x * blockDim.x + threadIdx.x;
  int stride = gridDim.x * blockDim.x;
  for (; i < n; i += stride) dst[i] = (bf16)src[i];
}

// ---------------- GEMM + bias: C[M,N] = alpha*(A[M,K] @ B[K,N] + bias) ----------
// M=4096, N=1024, K=1024. 256 threads = 8 waves. WG tile 128x128.
// wave w -> rows w*16..w*16+15, all 128 cols (8 col tiles). K staged 64-wide.
template <typename OutT>
__global__ void __launch_bounds__(256, 1)
gemm_bias_kernel(const bf16* __restrict__ A, const bf16* __restrict__ Bw,
                 const float* __restrict__ bias, OutT* __restrict__ C,
                 float alpha) {
  constexpr int N = DIMC;
  constexpr int K = DIMC;
  __shared__ bf16 As[128 * 64];   // row-major [128][64] for the current K-chunk (16 KB)
  __shared__ bf16 BT[128 * 64];   // transposed [n][k] for the current K-chunk  (16 KB)

  const int tid  = threadIdx.x;
  const int wave = tid >> 5;
  const int lane = tid & 31;
  const int g = lane >> 4;
  const int r = lane & 15;
  const int row0 = blockIdx.x * 128;
  const int col0 = blockIdx.y * 128;

  v8f acc[8];
#pragma unroll
  for (int ct = 0; ct < 8; ++ct) acc[ct] = zero8();

  for (int kc0 = 0; kc0 < K; kc0 += 64) {
    __syncthreads();
    // stage A tile: 128x64 bf16 (coalesced v8bf loads, 4 per thread)
    for (int it = tid; it < 1024; it += 256) {
      const int rr = it >> 3;
      const int cc = (it & 7) * 8;
      *reinterpret_cast<v8bf*>(&As[rr * 64 + cc]) =
          *reinterpret_cast<const v8bf*>(&A[(size_t)(row0 + rr) * K + kc0 + cc]);
    }
    // stage B tile 64x128 transposed -> BT[n][k]: one v8bf global load per it,
    // 8 strided LDS stores (transpose scatter)
    for (int it = tid; it < 1024; it += 256) {
      const int kk = it >> 4;          // 0..63
      const int nn = (it & 15) * 8;    // 0,8,...,120
      const v8bf vv =
          *reinterpret_cast<const v8bf*>(&Bw[(size_t)(kc0 + kk) * N + col0 + nn]);
#pragma unroll
      for (int e = 0; e < 8; ++e) BT[(nn + e) * 64 + kk] = vv[e];
    }
    __syncthreads();

#pragma unroll
    for (int kc = 0; kc < 2; ++kc) {
      const v16bf a = load_a_frag(&As[(wave * 16 + r) * 64], g, kc);
#pragma unroll
      for (int ct = 0; ct < 8; ++ct) {
        const v16bf b = load_b_frag(&BT[(ct * 16 + r) * 64 + kc * 32 + g * 16]);
        acc[ct] = wmma_bf16(a, b, acc[ct]);
      }
    }
  }

  // epilogue: D layout -> row = j + 8g, col = r (within 16x16 tile)
#pragma unroll
  for (int ct = 0; ct < 8; ++ct) {
#pragma unroll
    for (int j = 0; j < 8; ++j) {
      const int rr = row0 + wave * 16 + 8 * g + j;
      const int cc = col0 + ct * 16 + r;
      const float v = (acc[ct][j] + bias[cc]) * alpha;
      C[(size_t)rr * N + cc] = (OutT)v;
    }
  }
}

// ---------------- block-sparse flash attention ----------------
// grid = (NB, HEADS, BATCH), 128 threads = 4 waves.
// Per the reference's raw reshape, head h of batch b is the contiguous
// [SEQ x HD] slab at offset b*SEQ*DIM + h*SEQ*HD of the projection output.
// Q is pre-scaled by 1/sqrt(HD) at projection time.
__global__ void __launch_bounds__(128, 1)
attn_kernel(const bf16* __restrict__ Qb, const bf16* __restrict__ Kb,
            const bf16* __restrict__ Vb, const int* __restrict__ rsrc,
            const int* __restrict__ rtgt, bf16* __restrict__ Ab) {
  __shared__ bf16 Qs[BLKC * HDC];       // [64][128] row-major  (16 KB)
  __shared__ bf16 Ks[BLKC * HDC];       // [64][128] row-major  (16 KB)
  __shared__ bf16 VT[HDC * BLKC];       // [128][64] = V^T      (16 KB)
  __shared__ bf16 Ps[4 * 16 * BLKC];    // per-wave P tile      ( 8 KB)

  const int qb = blockIdx.x;
  const int h  = blockIdx.y;
  const int b  = blockIdx.z;
  const int tid  = threadIdx.x;
  const int wave = tid >> 5;
  const int lane = tid & 31;
  const int g = lane >> 4;
  const int r = lane & 15;

  const size_t headoff = (size_t)b * SEQC * DIMC + (size_t)h * SEQC * HDC;
  const bf16* Qh = Qb + headoff;  // [2048][128]
  const bf16* Kh = Kb + headoff;
  const bf16* Vh = Vb + headoff;

  const int rs0 = rsrc[0], rs1 = rsrc[1], rs2 = rsrc[2];
  const int rt0 = rtgt[0], rt1 = rtgt[1], rt2 = rtgt[2];

  // stage the 64x128 Q tile once (visible after the first in-loop barrier)
  for (int it = tid; it < BLKC * 16; it += 128) {
    const int rr = it >> 4;
    const int cc = (it & 15) * 8;
    *reinterpret_cast<v8bf*>(&Qs[rr * HDC + cc]) =
        *reinterpret_cast<const v8bf*>(&Qh[(size_t)(qb * BLKC + rr) * HDC + cc]);
  }
  const bf16* qsrow = &Qs[(wave * 16 + r) * HDC];

  v8f o[8];
#pragma unroll
  for (int ct = 0; ct < 8; ++ct) o[ct] = zero8();
  float m_run[8], l_run[8];
#pragma unroll
  for (int j = 0; j < 8; ++j) { m_run[j] = -1e30f; l_run[j] = 0.0f; }

  for (int kb = 0; kb < NBC; ++kb) {
    const bool allowed = (kb == 0) || (kb == NBC - 1) ||
                         (kb >= qb - 1 && kb <= qb + 1) ||
                         (rs0 == qb && rt0 == kb) ||
                         (rs1 == qb && rt1 == kb) ||
                         (rs2 == qb && rt2 == kb);
    if (!allowed) continue;  // uniform across the block

    __syncthreads();
    // stage K block row-major (coalesced v8bf)
    for (int it = tid; it < BLKC * 16; it += 128) {
      const int rr = it >> 4;
      const int cc = (it & 15) * 8;
      *reinterpret_cast<v8bf*>(&Ks[rr * HDC + cc]) =
          *reinterpret_cast<const v8bf*>(&Kh[(size_t)(kb * BLKC + rr) * HDC + cc]);
    }
    // stage V transposed: one v8bf global load per it, 8 strided LDS stores
    for (int it = tid; it < BLKC * 16; it += 128) {
      const int kk = it >> 4;          // 0..63
      const int dd = (it & 15) * 8;    // 0,8,...,120
      const v8bf vv =
          *reinterpret_cast<const v8bf*>(&Vh[(size_t)(kb * BLKC + kk) * HDC + dd]);
#pragma unroll
      for (int e = 0; e < 8; ++e) VT[(dd + e) * BLKC + kk] = vv[e];
    }
    __syncthreads();

    // S = Q @ K^T : 16x64 per wave, K-dim 128 (Q already has 1/sqrt(HD) folded in)
    v8f s[4];
#pragma unroll
    for (int ct = 0; ct < 4; ++ct) s[ct] = zero8();
#pragma unroll
    for (int kc = 0; kc < 4; ++kc) {
      const v16bf aq = load_a_frag(qsrow, g, kc);
#pragma unroll
      for (int ct = 0; ct < 4; ++ct) {
        const v16bf bk = load_b_frag(&Ks[(ct * 16 + r) * HDC + kc * 32 + g * 16]);
        s[ct] = wmma_bf16(aq, bk, s[ct]);
      }
    }

    // row max over 64 cols: within-lane over 4 tiles, then across 16 lanes
    float mx[8];
#pragma unroll
    for (int j = 0; j < 8; ++j)
      mx[j] = fmaxf(fmaxf(s[0][j], s[1][j]), fmaxf(s[2][j], s[3][j]));
#pragma unroll
    for (int j = 0; j < 8; ++j) {
#pragma unroll
      for (int off = 1; off < 16; off <<= 1)
        mx[j] = fmaxf(mx[j], __shfl_xor(mx[j], off, 32));
    }

    float mnew[8], corr[8], rsum[8];
#pragma unroll
    for (int j = 0; j < 8; ++j) {
      mnew[j] = fmaxf(m_run[j], mx[j]);
      corr[j] = __expf(m_run[j] - mnew[j]);
      rsum[j] = 0.0f;
      m_run[j] = mnew[j];
    }

    // P = exp(S - mnew); C-layout (row j+8g, col ct*16+r) -> LDS (A-layout source)
    bf16* Pw = &Ps[wave * 16 * BLKC];
#pragma unroll
    for (int ct = 0; ct < 4; ++ct) {
#pragma unroll
      for (int j = 0; j < 8; ++j) {
        const float p = __expf(s[ct][j] - mnew[j]);
        rsum[j] += p;
        Pw[(j + 8 * g) * BLKC + ct * 16 + r] = (bf16)p;
      }
    }
#pragma unroll
    for (int j = 0; j < 8; ++j) {
#pragma unroll
      for (int off = 1; off < 16; off <<= 1)
        rsum[j] += __shfl_xor(rsum[j], off, 32);
      l_run[j] = l_run[j] * corr[j] + rsum[j];
    }
    // rescale O
#pragma unroll
    for (int ct = 0; ct < 8; ++ct)
#pragma unroll
      for (int j = 0; j < 8; ++j) o[ct][j] *= corr[j];

    // O += P @ V  (P: 16x64 from wave-private LDS; V^T gives contiguous B frags)
    const bf16* Prow = &Pw[r * BLKC];
#pragma unroll
    for (int kcc = 0; kcc < 2; ++kcc) {
      const v16bf ap = load_a_frag(Prow, g, kcc);
#pragma unroll
      for (int ct = 0; ct < 8; ++ct) {
        const v16bf bv = load_b_frag(&VT[(ct * 16 + r) * BLKC + kcc * 32 + g * 16]);
        o[ct] = wmma_bf16(ap, bv, o[ct]);
      }
    }
  }

  // normalize and write out in [B, S, H*HD] layout (transpose(1,2) + reshape)
#pragma unroll
  for (int ct = 0; ct < 8; ++ct) {
#pragma unroll
    for (int j = 0; j < 8; ++j) {
      const int srow = qb * BLKC + wave * 16 + 8 * g + j;
      const int d = ct * 16 + r;
      const float val = o[ct][j] / l_run[j];
      Ab[(size_t)b * SEQC * DIMC + (size_t)srow * DIMC + h * HDC + d] = (bf16)val;
    }
  }
}

// ---------------- host entry ----------------
extern "C" void kernel_launch(void* const* d_in, const int* in_sizes, int n_in,
                              void* d_out, int out_size, void* d_ws, size_t ws_size,
                              hipStream_t stream) {
  const float* x  = (const float*)d_in[0];
  const float* Wq = (const float*)d_in[1];
  const float* bq = (const float*)d_in[2];
  const float* Wk = (const float*)d_in[3];
  const float* bk = (const float*)d_in[4];
  const float* Wv = (const float*)d_in[5];
  const float* bv = (const float*)d_in[6];
  const float* Wo = (const float*)d_in[7];
  const float* bo = (const float*)d_in[8];
  const int* rsrc = (const int*)d_in[9];
  const int* rtgt = (const int*)d_in[10];
  float* out = (float*)d_out;

  const int NX = BATCHC * SEQC * DIMC;  // 4,194,304
  const int NW = DIMC * DIMC;           // 1,048,576

  bf16* xb  = (bf16*)d_ws;
  bf16* Wqb = xb + NX;
  bf16* Wkb = Wqb + NW;
  bf16* Wvb = Wkb + NW;
  bf16* Wob = Wvb + NW;
  bf16* Qb  = Wob + NW;
  bf16* Kb  = Qb + NX;
  bf16* Vb  = Kb + NX;
  bf16* Ab  = Vb + NX;

  cvt_kernel<<<2048, 256, 0, stream>>>(x, xb, NX);
  cvt_kernel<<<512, 256, 0, stream>>>(Wq, Wqb, NW);
  cvt_kernel<<<512, 256, 0, stream>>>(Wk, Wkb, NW);
  cvt_kernel<<<512, 256, 0, stream>>>(Wv, Wvb, NW);
  cvt_kernel<<<512, 256, 0, stream>>>(Wo, Wob, NW);

  const float qscale = 0.088388347648318447f;  // 1/sqrt(HD)
  dim3 ggrid(32, 8);  // 4096/128 x 1024/128
  gemm_bias_kernel<bf16><<<ggrid, 256, 0, stream>>>(xb, Wqb, bq, Qb, qscale);
  gemm_bias_kernel<bf16><<<ggrid, 256, 0, stream>>>(xb, Wkb, bk, Kb, 1.0f);
  gemm_bias_kernel<bf16><<<ggrid, 256, 0, stream>>>(xb, Wvb, bv, Vb, 1.0f);

  attn_kernel<<<dim3(NBC, HEADSC, BATCHC), 128, 0, stream>>>(Qb, Kb, Vb, rsrc, rtgt, Ab);

  gemm_bias_kernel<float><<<ggrid, 256, 0, stream>>>(Ab, Wob, bo, out, 1.0f);
}
